// EncoderGNN_65532611002932
// MI455X (gfx1250) — matrix operand order repaired and checked by reading, not compiled, not run on hardware
//
#include <hip/hip_runtime.h>
#include <hip/hip_fp16.h>

typedef __attribute__((ext_vector_type(16))) _Float16 v16h;
typedef __attribute__((ext_vector_type(8)))  _Float16 v8h;
typedef __attribute__((ext_vector_type(4)))  _Float16 v4h;
typedef __attribute__((ext_vector_type(8)))  float    v8f;

#define HD 128          // hidden dim
#define PE 16           // output dim
#define ROWP (HD + 8)   // padded LDS row (halfs): 136 = 17*8, keeps 16B align

__device__ __forceinline__ float4 ld4(const float* p) { return *(const float4*)p; }

// ---- pack a row-major f32 weight matrix [128 x (ntiles*16)] into WMMA-B ----
// layout: p = ((tile*4 + ks)*32 + lane)*16 + i  <->  W[(ks*32 + (lane>>4)*8 + (i&7) + (i>>3)*16)*ld + tile*16 + (lane&15)]
__global__ void pack_b_k(const float* __restrict__ W, _Float16* __restrict__ out,
                         int ld, int total) {
  int o = blockIdx.x * blockDim.x + threadIdx.x;
  if (o >= total) return;
  int i    = o & 15;
  int lane = (o >> 4) & 31;
  int ks   = (o >> 9) & 3;
  int tile = o >> 11;
  int k    = ks * 32 + ((lane >> 4) << 3) + (i & 7) + ((i >> 3) << 4);
  int col  = tile * 16 + (lane & 15);
  out[o] = (_Float16)W[(size_t)k * ld + col];
}

// ---------------- node embedding: h = type_embed[t] + x_feat@featW + featb ----
__global__ void node_embed_k(const int* __restrict__ x_type,
                             const float* __restrict__ x_feat,
                             const float* __restrict__ type_embed,
                             const float* __restrict__ featW,
                             const float* __restrict__ featb,
                             _Float16* __restrict__ h, int N) {
  int gid = blockIdx.x * blockDim.x + threadIdx.x;
  int n = gid >> 5;
  if (n >= N) return;
  int c = (gid & 31) << 2;
  int t = x_type[n];
  float4 acc = ld4(&type_embed[t * HD + c]);
  float4 fb  = ld4(&featb[c]);
  acc.x += fb.x; acc.y += fb.y; acc.z += fb.z; acc.w += fb.w;
#pragma unroll
  for (int f = 0; f < 7; ++f) {
    float xf = x_feat[n * 7 + f];
    float4 w = ld4(&featW[f * HD + c]);
    acc.x += xf * w.x; acc.y += xf * w.y; acc.z += xf * w.z; acc.w += xf * w.w;
  }
  v4h o = {(_Float16)acc.x, (_Float16)acc.y, (_Float16)acc.z, (_Float16)acc.w};
  *(v4h*)&h[(size_t)n * HD + c] = o;
}

// ---------------- agg = (1+eps[l]) * h  (folds the GIN self term) -------------
__global__ void init_agg_k(const _Float16* __restrict__ h,
                           _Float16* __restrict__ agg,
                           const float* __restrict__ eps, int layer, int N) {
  int gid = blockIdx.x * blockDim.x + threadIdx.x;
  int n = gid >> 5;
  if (n >= N) return;
  int c = (gid & 31) << 2;
  float s = 1.0f + eps[layer];
  v4h v = *(const v4h*)&h[(size_t)n * HD + c];
  v4h o = {(_Float16)(s * (float)v[0]), (_Float16)(s * (float)v[1]),
           (_Float16)(s * (float)v[2]), (_Float16)(s * (float)v[3])};
  *(v4h*)&agg[(size_t)n * HD + c] = o;
}

// ------- per-edge: agg[dst] += relu(h[src] + edge_attr@edgeW + edgeb) ---------
// Grid-stride over edges; edgeW/edgeb hoisted out of the loop; packed f16
// atomics halve the scatter traffic and atomic-op count.
__global__ void edge_msg_k(const _Float16* __restrict__ h,
                           const int* __restrict__ edge_index,
                           const float* __restrict__ edge_attr,
                           const float* __restrict__ edgeW,
                           const float* __restrict__ edgeb,
                           _Float16* __restrict__ agg, int E) {
  int gid  = blockIdx.x * blockDim.x + threadIdx.x;
  int c    = (gid & 31) << 2;
  int grp  = gid >> 5;                               // edge-group id
  int ngrp = (gridDim.x * blockDim.x) >> 5;
  float4 w0 = ld4(&edgeW[0 * HD + c]);
  float4 w1 = ld4(&edgeW[1 * HD + c]);
  float4 w2 = ld4(&edgeW[2 * HD + c]);
  float4 w3 = ld4(&edgeW[3 * HD + c]);
  float4 eb = ld4(&edgeb[c]);
  for (int e = grp; e < E; e += ngrp) {
    int s = edge_index[e];
    int d = edge_index[E + e];
    float4 ea = ld4(&edge_attr[(size_t)e * 4]);
    v4h hv = *(const v4h*)&h[(size_t)s * HD + c];
    float m0 = (float)hv[0] + eb.x + ea.x * w0.x + ea.y * w1.x + ea.z * w2.x + ea.w * w3.x;
    float m1 = (float)hv[1] + eb.y + ea.x * w0.y + ea.y * w1.y + ea.z * w2.y + ea.w * w3.y;
    float m2 = (float)hv[2] + eb.z + ea.x * w0.z + ea.y * w1.z + ea.z * w2.z + ea.w * w3.z;
    float m3 = (float)hv[3] + eb.w + ea.x * w0.w + ea.y * w1.w + ea.z * w2.w + ea.w * w3.w;
    m0 = m0 > 0.f ? m0 : 0.f;
    m1 = m1 > 0.f ? m1 : 0.f;
    m2 = m2 > 0.f ? m2 : 0.f;
    m3 = m3 > 0.f ? m3 : 0.f;
    __half2* dst = (__half2*)&agg[(size_t)d * HD + c];
    unsafeAtomicAdd(dst,     __floats2half2_rn(m0, m1));  // global_atomic_pk_add_f16
    unsafeAtomicAdd(dst + 1, __floats2half2_rn(m2, m3));
  }
}

// ----- per-node MLP: Hout = relu( relu(Z@W1+b1) @ W2 + b2 ) via WMMA f16 ------
// Block = 8 waves; block owns 16 rows; wave w owns cols [16w, 16w+16).
// Z (= agg) is f16; weights are pre-packed in WMMA-B layout.
__global__ void gin_mlp_k(const _Float16* __restrict__ Z,
                          const _Float16* __restrict__ W1p,
                          const float* __restrict__ b1,
                          const _Float16* __restrict__ W2p,
                          const float* __restrict__ b2,
                          _Float16* __restrict__ Hout, int N) {
  __shared__ __align__(16) _Float16 inter[16 * ROWP];
  const int lane = threadIdx.x & 31;
  const int wave = threadIdx.x >> 5;
  const int hlf  = lane >> 4;
  const int m    = lane & 15;
  const int rowBase = blockIdx.x * 16;
  int rowA = rowBase + m; if (rowA > N - 1) rowA = N - 1;
  const int col = wave * 16 + m;
  const int ka  = hlf * 8;

  // ---- GEMM 1: Z @ W1 ----
  v8f acc = {};
#pragma unroll
  for (int ks = 0; ks < 4; ++ks) {
    const int k0 = ks * 32;
    v8h alo = *(const v8h*)&Z[(size_t)rowA * HD + k0 + ka];
    v8h ahi = *(const v8h*)&Z[(size_t)rowA * HD + k0 + 16 + ka];
    v16h a = __builtin_shufflevector(alo, ahi, 0, 1, 2, 3, 4, 5, 6, 7,
                                     8, 9, 10, 11, 12, 13, 14, 15);
    v16h b = *(const v16h*)&W1p[(size_t)(((wave * 4 + ks) * 32 + lane) << 4)];
    acc = __builtin_amdgcn_wmma_f32_16x16x32_f16(false, a, false, b,
                                                 (short)0, acc, false, false);
  }
  const float bias1 = b1[col];
#pragma unroll
  for (int r = 0; r < 8; ++r) {
    float v = acc[r] + bias1;
    v = v > 0.f ? v : 0.f;
    inter[(r + hlf * 8) * ROWP + col] = (_Float16)v;   // D layout -> LDS tile
  }
  __syncthreads();

  // ---- GEMM 2: inter @ W2 ----
  v8f acc2 = {};
#pragma unroll
  for (int ks = 0; ks < 4; ++ks) {
    const int k0 = ks * 32;
    v8h alo = *(const v8h*)&inter[m * ROWP + k0 + ka];
    v8h ahi = *(const v8h*)&inter[m * ROWP + k0 + 16 + ka];
    v16h a = __builtin_shufflevector(alo, ahi, 0, 1, 2, 3, 4, 5, 6, 7,
                                     8, 9, 10, 11, 12, 13, 14, 15);
    v16h b = *(const v16h*)&W2p[(size_t)(((wave * 4 + ks) * 32 + lane) << 4)];
    acc2 = __builtin_amdgcn_wmma_f32_16x16x32_f16(false, a, false, b,
                                                  (short)0, acc2, false, false);
  }
  const float bias2 = b2[col];
#pragma unroll
  for (int r = 0; r < 8; ++r) {
    int row = rowBase + r + hlf * 8;
    if (row < N) {
      float v = acc2[r] + bias2;
      Hout[(size_t)row * HD + col] = (_Float16)(v > 0.f ? v : 0.f);
    }
  }
}

// ---------------- out = h @ Wout + bout  (128 -> 16) via WMMA -----------------
__global__ void out_proj_k(const _Float16* __restrict__ Hin,
                           const _Float16* __restrict__ Woutp,
                           const float* __restrict__ bout,
                           float* __restrict__ out, int N) {
  const int lane = threadIdx.x & 31;
  const int wave = threadIdx.x >> 5;
  const int hlf  = lane >> 4;
  const int m    = lane & 15;
  const int rowBase = blockIdx.x * 128 + wave * 16;
  int rowA = rowBase + m; if (rowA > N - 1) rowA = N - 1;
  const int ka = hlf * 8;

  v8f acc = {};
#pragma unroll
  for (int ks = 0; ks < 4; ++ks) {
    const int k0 = ks * 32;
    v8h alo = *(const v8h*)&Hin[(size_t)rowA * HD + k0 + ka];
    v8h ahi = *(const v8h*)&Hin[(size_t)rowA * HD + k0 + 16 + ka];
    v16h a = __builtin_shufflevector(alo, ahi, 0, 1, 2, 3, 4, 5, 6, 7,
                                     8, 9, 10, 11, 12, 13, 14, 15);
    v16h b = *(const v16h*)&Woutp[(size_t)((ks * 32 + lane) << 4)];
    acc = __builtin_amdgcn_wmma_f32_16x16x32_f16(false, a, false, b,
                                                 (short)0, acc, false, false);
  }
  const float bias = bout[m];
#pragma unroll
  for (int r = 0; r < 8; ++r) {
    int row = rowBase + r + hlf * 8;
    if (row < N) out[(size_t)row * PE + m] = acc[r] + bias;
  }
}

extern "C" void kernel_launch(void* const* d_in, const int* in_sizes, int n_in,
                              void* d_out, int out_size, void* d_ws, size_t ws_size,
                              hipStream_t stream) {
  const int*   x_type     = (const int*)d_in[0];
  const float* x_feat     = (const float*)d_in[1];
  const int*   edge_index = (const int*)d_in[2];
  const float* edge_attr  = (const float*)d_in[3];
  const float* type_embed = (const float*)d_in[4];
  const float* featW      = (const float*)d_in[5];
  const float* featb      = (const float*)d_in[6];
  const float* edgeW      = (const float*)d_in[7];
  const float* edgeb      = (const float*)d_in[8];
  const float* W1         = (const float*)d_in[9];
  const float* b1         = (const float*)d_in[10];
  const float* W2         = (const float*)d_in[11];
  const float* b2         = (const float*)d_in[12];
  const float* eps        = (const float*)d_in[13];
  const float* Wout       = (const float*)d_in[14];
  const float* bout       = (const float*)d_in[15];

  const int N = in_sizes[0];
  const int E = in_sizes[3] / 4;

  // workspace layout (all f16)
  _Float16* h     = (_Float16*)d_ws;               // [N,128]
  _Float16* agg   = h + (size_t)N * HD;            // [N,128]
  _Float16* W1p   = agg + (size_t)N * HD;          // 3 x 16384
  _Float16* W2p   = W1p + 3 * HD * HD;             // 3 x 16384
  _Float16* Woutp = W2p + 3 * HD * HD;             // 2048

  dim3 blk(256);
  const int gNode = (N * 32 + 255) / 256;
  const int gEdge = 4096;                          // grid-stride over edges
  const int gMlp  = (N + 15) / 16;
  const int gOut  = (N + 127) / 128;

  // pre-pack weights into WMMA-B layout (f16)
  for (int l = 0; l < 3; ++l) {
    pack_b_k<<<64, blk, 0, stream>>>(W1 + (size_t)l * HD * HD,
                                     W1p + (size_t)l * HD * HD, HD, HD * HD);
    pack_b_k<<<64, blk, 0, stream>>>(W2 + (size_t)l * HD * HD,
                                     W2p + (size_t)l * HD * HD, HD, HD * HD);
  }
  pack_b_k<<<8, blk, 0, stream>>>(Wout, Woutp, PE, HD * PE);

  node_embed_k<<<gNode, blk, 0, stream>>>(x_type, x_feat, type_embed, featW,
                                          featb, h, N);
  for (int l = 0; l < 3; ++l) {
    init_agg_k<<<gNode, blk, 0, stream>>>(h, agg, eps, l, N);
    edge_msg_k<<<gEdge, blk, 0, stream>>>(h, edge_index, edge_attr, edgeW,
                                          edgeb, agg, E);
    gin_mlp_k<<<gMlp, blk, 0, stream>>>(agg, W1p + (size_t)l * HD * HD,
                                        b1 + (size_t)l * HD,
                                        W2p + (size_t)l * HD * HD,
                                        b2 + (size_t)l * HD, h, N);
  }
  out_proj_k<<<gOut, blk, 0, stream>>>(h, Woutp, bout, (float*)d_out, N);
}
